// AttnDecoderLSTM_29016799051922
// MI455X (gfx1250) — compile-verified, matrix operand order checked by reading, not run on hardware
//
#include <hip/hip_runtime.h>
#include <hip/hip_bf16.h>

// Problem dims (fixed by the reference)
#define SEQ   512
#define BATCH 256
#define HID   512

typedef __attribute__((ext_vector_type(16))) __bf16 v16bf;
typedef __attribute__((ext_vector_type(8)))  float  v8f;

// ---------------------------------------------------------------------------
// WMMA helpers (CDNA5 wave32, V_WMMA_F32_16X16X32_BF16)
// A fragment layout per ISA "16-bit A-Matrix 16x32":
//   lane m = lane&15, half = lane>>4
//   VGPR j holds K pair base = 2*(j&3) + 16*(j>>2) + 8*half
// B fragment uses the transposed-identical layout loaded from row-major [n][k].
// C/D: VGPR r -> M = r + 8*half, N = lane&15.
// ---------------------------------------------------------------------------
__device__ __forceinline__ int frag_k(int j, int half) {
  return 2 * (j & 3) + 16 * (j >> 2) + 8 * half;
}

__device__ __forceinline__ v8f v8f_zero() {
  v8f z;
#pragma unroll
  for (int i = 0; i < 8; ++i) z[i] = 0.0f;
  return z;
}

// Load a 16x32 bf16 fragment from a row-major LDS tile [16][stride] (bf16).
__device__ __forceinline__ v16bf load_frag_lds(const __bf16* tile, int stride,
                                               int kbase, int lane) {
  const int row = lane & 15, half = lane >> 4;
  v16bf f;
#pragma unroll
  for (int j = 0; j < 8; ++j) {
    const int k = kbase + frag_k(j, half);
    f[2 * j]     = tile[row * stride + k];
    f[2 * j + 1] = tile[row * stride + k + 1];
  }
  return f;
}

// Load a 16x32 bf16 fragment from a row-major *fp32* LDS tile with the
// TDM-pad layout: 1 extra DWORD inserted after every 256 DWORDs (row stride
// = 514 DW for 512-elem rows). Compensation: dword_off = k + (k>>8).
__device__ __forceinline__ v16bf load_frag_lds_f32(const float* tile, int stride,
                                                   int kbase, int lane) {
  const int row = lane & 15, half = lane >> 4;
  v16bf f;
#pragma unroll
  for (int j = 0; j < 8; ++j) {
    const int k = kbase + frag_k(j, half);
    const int koff = k + (k >> 8);  // TDM pad compensation
    const float2 p = *(const float2*)(tile + row * stride + koff);
    f[2 * j]     = (__bf16)p.x;
    f[2 * j + 1] = (__bf16)p.y;
  }
  return f;
}

// Load a 16x32 bf16 fragment from a row-major fp32 global matrix [16 rows][ldk],
// converting fp32 -> bf16 on the fly (pairs contiguous in k -> b128 loads).
__device__ __forceinline__ v16bf load_frag_global_f32(const float* __restrict__ base,
                                                      int ldk, int kbase, int lane) {
  const int row = lane & 15, half = lane >> 4;
  v16bf f;
#pragma unroll
  for (int j = 0; j < 8; ++j) {
    const int k = kbase + frag_k(j, half);
    const float2 p = *(const float2*)(base + (size_t)row * ldk + k);
    f[2 * j]     = (__bf16)p.x;
    f[2 * j + 1] = (__bf16)p.y;
  }
  return f;
}

__device__ __forceinline__ v8f wmma_bf16(v16bf a, v16bf b, v8f c) {
  // (neg_a, A, neg_b, B, c_mod, C, reuse_a, reuse_b)
  return __builtin_amdgcn_wmma_f32_16x16x32_bf16(false, a, false, b, (short)0, c,
                                                 false, false);
}

// ---------------------------------------------------------------------------
// Tensor Data Mover: 2-D fp32 tile load (global -> LDS) with LDS padding of
// 1 DWORD per 256 DWORDs (so 512-elem rows land at a 514-DW stride, breaking
// LDS bank conflicts for the WMMA fragment loads).
// D# built per CDNA5 ISA ch.8: group0 = {count,lds_addr,global_addr,type=2},
// group1 = {data_size=4B, pad_enable, pad_interval=256DW, pad_amount=1DW,
//           tensor_dim0/1, tile_dim0/1, tensor_dim0_stride}.
// ---------------------------------------------------------------------------
#if defined(__gfx1250__) && __has_builtin(__builtin_amdgcn_tensor_load_to_lds) && \
    __has_builtin(__builtin_amdgcn_s_wait_tensorcnt)
#define HAVE_TDM 1
#else
#define HAVE_TDM 0
#endif

#if HAVE_TDM
typedef unsigned int u32x4 __attribute__((ext_vector_type(4)));
typedef int          i32x4 __attribute__((ext_vector_type(4)));
typedef int          i32x8 __attribute__((ext_vector_type(8)));

__device__ __forceinline__ void tdm_load_2d_f32(const float* gsrc, unsigned lds_byte_addr,
                                                unsigned tile_w /*elems*/, unsigned tile_h,
                                                unsigned long long row_stride /*elems*/) {
  const unsigned long long ga = (unsigned long long)(size_t)gsrc;
  u32x4 g0;
  g0[0] = 1u;                                   // count=1 (valid user descriptor)
  g0[1] = lds_byte_addr;                        // lds_addr [63:32]
  g0[2] = (unsigned)(ga & 0xffffffffu);         // global_addr[31:0]
  g0[3] = (unsigned)((ga >> 32) & 0x1ffffffu)   // global_addr[56:32]
        | (2u << 30);                           // type = 2 ("image")
  i32x8 g1;
  g1[0] = (2 << 16)                             // data_size = 4B
        | (1 << 20)                             // pad_enable
        | (7 << 22)                             // pad_interval: 256 DWORDs
        | (0 << 25);                            // pad_amount: 1 DWORD
  g1[1] = (int)((tile_w & 0xffffu) << 16);      // tensor_dim0[15:0]
  g1[2] = (int)((tile_w >> 16) & 0xffffu)       // tensor_dim0[31:16]
        | (int)((tile_h & 0xffffu) << 16);      // tensor_dim1[15:0]
  g1[3] = (int)((tile_w & 0xffffu) << 16);      // tensor_dim1[31:16]=0 | tile_dim0
  g1[4] = (int)(tile_h & 0xffffu);              // tile_dim1 (tile_dim2 = 0)
  g1[5] = (int)(unsigned)(row_stride & 0xffffffffu);        // dim0_stride[31:0]
  g1[6] = (int)(unsigned)((row_stride >> 32) & 0xffffu);    // dim0_stride[47:32]
  g1[7] = 0;
  const i32x4 z4 = {0, 0, 0, 0};
#if __clang_major__ <= 22
  __builtin_amdgcn_tensor_load_to_lds(g0, g1, z4, z4, 0);               // ROCm 7.2 arity
#else
  const i32x8 z8 = {0, 0, 0, 0, 0, 0, 0, 0};
  __builtin_amdgcn_tensor_load_to_lds(g0, g1, z4, z4, z8, 0);           // clang-23 arity
#endif
}
#endif  // HAVE_TDM

// ---------------------------------------------------------------------------
// Fused attention kernel.
// One workgroup (8 waves, 256 threads) handles 16 s-rows at a fixed batch b.
// LDS (dynamic, ~124KB; <=320KB/WGP on CDNA5, allows 2 WGs/WGP):
//   Q  : 16 x 1024 bf16 (padded)  -> later reused as y = [h | attn_applied]
//   L  : 16 x 512  f32  logits    (aliased over VT region; disjoint lifetime)
//   VT : 512 x 64  bf16 V^T chunk (aliased with L)
//   P  : 16 x 512  bf16 softmax probs
// ---------------------------------------------------------------------------
#define QS 1032   // Q row stride (bf16), padded vs 1024 to break bank conflicts
#define LS 516    // logits row stride (f32)
#define PS 520    // P row stride (bf16)
#define VS 72     // VT row stride (bf16): 64 t + pad

#define OFF_Q   0
#define OFF_VT  (16 * QS * 2)                 // 33024
#define OFF_L   OFF_VT                        // aliased (disjoint lifetimes)
#define OFF_P   (OFF_VT + 512 * VS * 2)       // 33024 + 73728
#define OFF_RED (OFF_P + 16 * PS * 2)         // + 16640
#define SMEM_BYTES (OFF_RED + 16 * 16 * 4)    // ~124 KB

__global__ __launch_bounds__(256) void attn_fused(
    const float* __restrict__ h, const float* __restrict__ enc,
    const float* __restrict__ W_attn, const float* __restrict__ b_attn,
    const float* __restrict__ W_comb, const float* __restrict__ b_comb,
    float* __restrict__ out_att) {
  extern __shared__ __align__(16) unsigned char smem[];
  __bf16* Q   = (__bf16*)(smem + OFF_Q);
  float*  L   = (float*)(smem + OFF_L);
  __bf16* VT  = (__bf16*)(smem + OFF_VT);
  __bf16* P   = (__bf16*)(smem + OFF_P);
  float*  red = (float*)(smem + OFF_RED);

  const int blk  = blockIdx.x;
  const int sblk = blk & 31;    // same-b blocks adjacent -> V shares L2
  const int b    = blk >> 5;
  const int s0   = sblk * 16;
  const int tid  = threadIdx.x;
  const int lane = tid & 31;
  const int wave = tid >> 5;

  // ---- Stage 1: stage Q = [h | enc] rows (s0..s0+15, b) into LDS as bf16 ----
  for (int i = tid; i < 16 * 1024; i += 256) {
    const int r = i >> 10, k = i & 1023;
    const float v = (k < 512) ? h[((size_t)(s0 + r) * BATCH + b) * HID + k]
                              : enc[((size_t)(s0 + r) * BATCH + b) * HID + (k - 512)];
    Q[r * QS + k] = (__bf16)v;
  }
  __syncthreads();

  // ---- Stage 2: logits tile per wave: t-cols [wave*64, wave*64+64) ----
  {
    const int t0 = wave * 64;
    v8f acc[4];
#pragma unroll
    for (int nt = 0; nt < 4; ++nt) acc[nt] = v8f_zero();
    for (int k = 0; k < 1024; k += 32) {
      if (k + 64 < 1024)  // prefetch W two k-chunks ahead (global_prefetch_b8)
        __builtin_prefetch(W_attn + (size_t)(t0 + (lane & 15)) * 1024 + k + 64, 0, 0);
      const v16bf a = load_frag_lds(Q, QS, k, lane);
#pragma unroll
      for (int nt = 0; nt < 4; ++nt) {
        const v16bf bfrag =
            load_frag_global_f32(W_attn + (size_t)(t0 + nt * 16) * 1024, 1024, k, lane);
        acc[nt] = wmma_bf16(a, bfrag, acc[nt]);
      }
    }
    const int n = lane & 15, half = lane >> 4;
#pragma unroll
    for (int nt = 0; nt < 4; ++nt) {
      const int t = t0 + nt * 16 + n;
      const float bias = b_attn[t];
#pragma unroll
      for (int r = 0; r < 8; ++r) {
        const int m = r + 8 * half;
        L[m * LS + t] = acc[nt][r] + bias;
      }
    }
  }
  __syncthreads();

  // ---- Stage 3: row softmax of L (16 rows x 512), write P as bf16 ----
  {
    const int row = tid & 15;
    const int ch  = tid >> 4;  // 16 chunks of 32 columns
    float mx = -1e30f;
    for (int c = ch * 32; c < ch * 32 + 32; ++c) mx = fmaxf(mx, L[row * LS + c]);
    red[ch * 16 + row] = mx;
    __syncthreads();
    float rmax = -1e30f;
    for (int c = 0; c < 16; ++c) rmax = fmaxf(rmax, red[c * 16 + row]);
    float s = 0.0f;
    for (int c = ch * 32; c < ch * 32 + 32; ++c) {
      const float e = __expf(L[row * LS + c] - rmax);
      L[row * LS + c] = e;
      s += e;
    }
    __syncthreads();  // all rmax reads of red done before overwrite
    red[ch * 16 + row] = s;
    __syncthreads();
    float rsum = 0.0f;
    for (int c = 0; c < 16; ++c) rsum += red[c * 16 + row];
    const float inv = 1.0f / rsum;
    for (int c = ch * 32; c < ch * 32 + 32; ++c)
      P[row * PS + c] = (__bf16)(L[row * LS + c] * inv);
  }
  __syncthreads();

  // ---- Stage 4: attn_applied tile = P(16x512) @ V(512x512), V[t][hh]=enc[t][b][hh]
  //      Stream V in 64-t chunks transposed into LDS (VT aliases dead L). ----
  v8f av[4];
#pragma unroll
  for (int nt = 0; nt < 4; ++nt) av[nt] = v8f_zero();
  for (int tc = 0; tc < 512; tc += 64) {
    __syncthreads();  // previous chunk fully consumed / P writes visible
    for (int i = tid; i < 64 * 512; i += 256) {
      const int tl = i >> 9, hh = i & 511;
      VT[hh * VS + tl] = (__bf16)enc[((size_t)(tc + tl) * BATCH + b) * HID + hh];
    }
    __syncthreads();
    const int h0 = wave * 64;
#pragma unroll
    for (int kk = 0; kk < 64; kk += 32) {
      const v16bf a = load_frag_lds(P, PS, tc + kk, lane);
#pragma unroll
      for (int nt = 0; nt < 4; ++nt) {
        const v16bf bfrag = load_frag_lds(VT + (h0 + nt * 16) * VS, VS, kk, lane);
        av[nt] = wmma_bf16(a, bfrag, av[nt]);
      }
    }
  }
  __syncthreads();

  // Write attn tile into Q cols [512,1024) -> Q becomes y = [h | attn_applied]
  {
    const int n = lane & 15, half = lane >> 4;
#pragma unroll
    for (int nt = 0; nt < 4; ++nt)
#pragma unroll
      for (int r = 0; r < 8; ++r) {
        const int m = r + 8 * half;
        Q[m * QS + 512 + wave * 64 + nt * 16 + n] = (__bf16)av[nt][r];
      }
  }
  __syncthreads();

  // ---- Stage 5: attended = y(16x1024) @ W_comb^T + b_comb ----
  {
    const int h0 = wave * 64;
    v8f oc[4];
#pragma unroll
    for (int nt = 0; nt < 4; ++nt) oc[nt] = v8f_zero();
    for (int k = 0; k < 1024; k += 32) {
      if (k + 64 < 1024)
        __builtin_prefetch(W_comb + (size_t)(h0 + (lane & 15)) * 1024 + k + 64, 0, 0);
      const v16bf a = load_frag_lds(Q, QS, k, lane);
#pragma unroll
      for (int nt = 0; nt < 4; ++nt) {
        const v16bf bfrag =
            load_frag_global_f32(W_comb + (size_t)(h0 + nt * 16) * 1024, 1024, k, lane);
        oc[nt] = wmma_bf16(a, bfrag, oc[nt]);
      }
    }
    const int n = lane & 15, half = lane >> 4;
#pragma unroll
    for (int nt = 0; nt < 4; ++nt) {
      const int hh = h0 + nt * 16 + n;
      const float bias = b_comb[hh];
#pragma unroll
      for (int r = 0; r < 8; ++r) {
        const int m = r + 8 * half;
        out_att[((size_t)(s0 + m) * BATCH + b) * HID + hh] = oc[nt][r] + bias;
      }
    }
  }
}

// ---------------------------------------------------------------------------
// One LSTM timestep. Grid: (BATCH/16) x (HID/64) = 128 WGs of 256 threads.
// A-tile ([x_t | h_prev], fp32) is staged into LDS by the Tensor Data Mover
// (two 2-D descriptors, one per source block; TENSORcnt + barrier), with TDM
// padding giving a conflict-free 514-DW row stride. Fragments convert
// fp32->bf16 at load (v_cvt_pk_bf16_f32) and feed v_wmma_f32_16x16x32_bf16.
// Recurrent h is double-buffered (hin/hout) to avoid cross-block RAW races.
// ---------------------------------------------------------------------------
#define ASW 514                                   // fp32 DW row stride (512 + 2 pads)
#define LSTM_AB  (16 * ASW * 4)                   // one 16x512 fp32 tile (padded)
#define LSTM_G_OFF (2 * LSTM_AB + 32)             // slack for trailing TDM pad
#define LSTM_SMEM  (LSTM_G_OFF + 4 * 16 * 64 * 4)

__global__ __launch_bounds__(256) void lstm_step(
    const float* __restrict__ x,      // h[t] : [BATCH][HID]
    const float* __restrict__ W_ih,   // [4H][H]
    const float* __restrict__ W_hh,   // [4H][H]
    const float* __restrict__ b_ih, const float* __restrict__ b_hh,
    const float* __restrict__ hin,    // [BATCH][HID]
    float* __restrict__ hout,         // [BATCH][HID]
    float* __restrict__ cstate,       // [BATCH][HID]
    float* __restrict__ out) {        // decoder_out[t]
  extern __shared__ __align__(16) unsigned char smem[];
  float* Af = (float*)smem;                       // x-tile   [16][ASW]
  float* Ah = Af + 16 * ASW;                      // h-tile   [16][ASW]
  float* G  = (float*)(smem + LSTM_G_OFF);        // gates [4][16][64]

  const int blk  = blockIdx.x;
  const int b0   = (blk & 15) * 16;
  const int h0   = (blk >> 4) * 64;
  const int tid  = threadIdx.x;
  const int lane = tid & 31;
  const int wave = tid >> 5;

#if HAVE_TDM
  if (wave == 0) {  // one wave issues both DMAs; EXEC ignored by TDM
    const unsigned lbase = (unsigned)(size_t)(void*)Af;  // LDS byte offset
    tdm_load_2d_f32(x   + (size_t)b0 * HID, lbase,           512, 16, 512);
    tdm_load_2d_f32(hin + (size_t)b0 * HID, lbase + LSTM_AB, 512, 16, 512);
    __builtin_amdgcn_s_wait_tensorcnt(0);
  }
  __syncthreads();
#else
  for (int i = tid; i < 16 * 1024; i += 256) {
    const int r = i >> 10, k = i & 1023;
    if (k < 512) {
      Af[r * ASW + k + (k >> 8)] = x[(size_t)(b0 + r) * HID + k];
    } else {
      const int kk = k - 512;
      Ah[r * ASW + kk + (kk >> 8)] = hin[(size_t)(b0 + r) * HID + kk];
    }
  }
  __syncthreads();
#endif

  // GEMM: gates[b][n] = sum_k A[b][k] * (k<512 ? W_ih : W_hh)[n][k']
  {
    const int g   = wave >> 1;          // gate 0..3 (i,f,g,o)
    const int hl0 = (wave & 1) * 32;    // 32-wide h sub-chunk
    v8f acc[2];
#pragma unroll
    for (int nt = 0; nt < 2; ++nt) acc[nt] = v8f_zero();
    for (int k = 0; k < 1024; k += 32) {
      const float* At; const float* Wb; int kk;
      if (k < 512) { At = Af; Wb = W_ih; kk = k; }
      else         { At = Ah; Wb = W_hh; kk = k - 512; }
      const v16bf a = load_frag_lds_f32(At, ASW, kk, lane);
#pragma unroll
      for (int nt = 0; nt < 2; ++nt) {
        const int n0 = g * 512 + h0 + hl0 + nt * 16;  // W row (gate index)
        const v16bf bfrag = load_frag_global_f32(Wb + (size_t)n0 * 512, 512, kk, lane);
        acc[nt] = wmma_bf16(a, bfrag, acc[nt]);
      }
    }
    const int n = lane & 15, half = lane >> 4;
#pragma unroll
    for (int nt = 0; nt < 2; ++nt)
#pragma unroll
      for (int r = 0; r < 8; ++r) {
        const int m = r + 8 * half;
        G[(g * 16 + m) * 64 + hl0 + nt * 16 + n] = acc[nt][r];
      }
  }
  __syncthreads();

  // Fused elementwise LSTM update (PyTorch gate order i,f,g,o)
  for (int e = tid; e < 16 * 64; e += 256) {
    const int m = e >> 6, hl = e & 63;
    const int bb = b0 + m, hh = h0 + hl;
    const float gi = G[(0 * 16 + m) * 64 + hl] + b_ih[0 * 512 + hh] + b_hh[0 * 512 + hh];
    const float gf = G[(1 * 16 + m) * 64 + hl] + b_ih[1 * 512 + hh] + b_hh[1 * 512 + hh];
    const float gg = G[(2 * 16 + m) * 64 + hl] + b_ih[2 * 512 + hh] + b_hh[2 * 512 + hh];
    const float go = G[(3 * 16 + m) * 64 + hl] + b_ih[3 * 512 + hh] + b_hh[3 * 512 + hh];
    const float i_ = 1.0f / (1.0f + __expf(-gi));
    const float f_ = 1.0f / (1.0f + __expf(-gf));
    const float o_ = 1.0f / (1.0f + __expf(-go));
    const float g_ = tanhf(gg);
    const size_t idx = (size_t)bb * HID + hh;
    const float c = f_ * cstate[idx] + i_ * g_;
    cstate[idx] = c;
    const float hv = o_ * tanhf(c);
    hout[idx] = hv;
    out[idx]  = hv;
  }
}

__global__ void zero_kernel(float* __restrict__ p, size_t n) {
  const size_t i = (size_t)blockIdx.x * blockDim.x + threadIdx.x;
  if (i < n) p[i] = 0.0f;
}

// ---------------------------------------------------------------------------
extern "C" void kernel_launch(void* const* d_in, const int* in_sizes, int n_in,
                              void* d_out, int out_size, void* d_ws, size_t ws_size,
                              hipStream_t stream) {
  (void)in_sizes; (void)n_in; (void)out_size; (void)ws_size;
  const float* h      = (const float*)d_in[0];
  const float* enc    = (const float*)d_in[1];
  const float* W_attn = (const float*)d_in[2];
  const float* b_attn = (const float*)d_in[3];
  const float* W_comb = (const float*)d_in[4];
  const float* b_comb = (const float*)d_in[5];
  const float* W_ih   = (const float*)d_in[6];
  const float* W_hh   = (const float*)d_in[7];
  const float* b_ih   = (const float*)d_in[8];
  const float* b_hh   = (const float*)d_in[9];

  float* dec_out = (float*)d_out;                          // [S][B][H]
  float* att_out = dec_out + (size_t)SEQ * BATCH * HID;    // [S][B][H]

  // Workspace: double-buffered recurrent h + cell state (1.5 MB)
  float* hA = (float*)d_ws;
  float* hB = hA + (size_t)BATCH * HID;
  float* cS = hB + (size_t)BATCH * HID;

  const size_t state_n = (size_t)3 * BATCH * HID;
  zero_kernel<<<(int)((state_n + 255) / 256), 256, 0, stream>>>(hA, state_n);

  // Fused attention: (S/16)*B = 8192 WGs, ~124KB dynamic LDS each.
  attn_fused<<<(SEQ / 16) * BATCH, 256, SMEM_BYTES, stream>>>(
      h, enc, W_attn, b_attn, W_comb, b_comb, att_out);

  // Sequential LSTM: one kernel per timestep, stream-ordered.
  for (int t = 0; t < SEQ; ++t) {
    const float* hin  = (t & 1) ? hB : hA;
    float*       hout = (t & 1) ? hA : hB;
    lstm_step<<<(BATCH / 16) * (HID / 64), 256, LSTM_SMEM, stream>>>(
        h + (size_t)t * BATCH * HID, W_ih, W_hh, b_ih, b_hh,
        hin, hout, cS, dec_out + (size_t)t * BATCH * HID);
  }
}
